// CPCUnsupersivedCriterion_59631325938114
// MI455X (gfx1250) — matrix operand order repaired and checked by reading, not compiled, not run on hardware
//
#include <hip/hip_runtime.h>

typedef __attribute__((ext_vector_type(16))) __bf16 v16bf;
typedef __attribute__((ext_vector_type(8)))  __bf16 v8bf;
typedef __attribute__((ext_vector_type(8)))  float  v8f;

#define B_   32
#define S_   128
#define D_   256
#define K_   12
#define NEG_ 128
#define W_   116           // S - K
#define BW_  (B_ * W_)     // 3712

// ---------------------------------------------------------------------------
// f32 -> bf16 conversion
// ---------------------------------------------------------------------------
__global__ void cvt_f32_bf16(const float* __restrict__ in, __bf16* __restrict__ out, int n) {
    int i = blockIdx.x * blockDim.x + threadIdx.x;
    if (i < n) out[i] = (__bf16)in[i];
}

__global__ void zero_accum(float* a) {
    if (threadIdx.x < 24) a[threadIdx.x] = 0.0f;
}

// A-fragment loader: one contiguous 16-elem run per lane (k = 16*half + s)
__device__ __forceinline__ v16bf load_afrag(const __bf16* aRow, int kc, int half) {
    v8bf alo = *(const v8bf*)(aRow + kc + half * 16);
    v8bf ahi = *(const v8bf*)(aRow + kc + half * 16 + 8);
    return __builtin_shufflevector(alo, ahi,
            0,1,2,3,4,5,6,7,8,9,10,11,12,13,14,15);
}

// ---------------------------------------------------------------------------
// Phase 1: locC[bw][k][e] = sum_a c[bw][a] * W[k][e][a]
// A = c rows (M=bw tile of 16, Kdim=a=256), B[a][e] = W[k][e][a] (contig in a)
// grid = 232 blocks (bw tiles), 256 threads = 8 waves; each wave walks (k,eTile)
// Pipelined: all 16 B b128-loads issued as a clause, then 8-deep WMMA chain.
// ---------------------------------------------------------------------------
__global__ __launch_bounds__(256) void locc_gemm(const __bf16* __restrict__ cB,
                                                 const __bf16* __restrict__ wB,
                                                 __bf16* __restrict__ locCb) {
    const int lane = threadIdx.x & 31;
    const int wid  = threadIdx.x >> 5;
    const int half = lane >> 4;       // K-group select
    const int l15  = lane & 15;
    const int bwTile = blockIdx.x * 16;

    // A row for this lane: global bw row = bwTile + (lane%16) -> cFeature[b][w<116]
    const int gbw = bwTile + l15;
    const int bb  = gbw / W_;
    const int ww  = gbw - bb * W_;
    const __bf16* aRow = cB + (size_t)(bb * S_ + ww) * D_;

    // Preload all 8 A fragments once; they are invariant across the tile loop.
    v16bf afr[8];
    #pragma unroll
    for (int s = 0; s < 8; ++s) afr[s] = load_afrag(aRow, s * 32, half);

    for (int t = wid; t < K_ * 16; t += 8) {
        const int k     = t >> 4;
        const int eTile = (t & 15) << 4;
        // B column = eTile + l15 ; K-run contiguous along a
        const __bf16* bRow = wB + ((size_t)(k * D_ + eTile + l15)) * D_;

        v16bf bfr[8];
        #pragma unroll
        for (int s = 0; s < 8; ++s)
            bfr[s] = *(const v16bf*)(bRow + s * 32 + half * 16);

        v8f acc = {};
        #pragma unroll
        for (int s = 0; s < 8; ++s)
            acc = __builtin_amdgcn_wmma_f32_16x16x32_bf16(
                      false, afr[s], false, bfr[s], (short)0, acc, false, false);

        // C/D layout: lane -> (N = lane%16, M = r + 8*(lane>=16))
        const int mBase = (lane >= 16) ? 8 : 0;
        #pragma unroll
        for (int r = 0; r < 8; ++r) {
            const int m = r + mBase;
            locCb[((size_t)((bwTile + m) * K_ + k)) * D_ + eTile + l15] = (__bf16)acc[r];
        }
    }
}

// ---------------------------------------------------------------------------
// Phase 2: per (b,w): negScore[k][n] = locC[k][:] . encB[extIdx(b,n,w)][:] / 256
//          -> 16x128 via 8 waves x (16x16, K=256) bf16 WMMA; rows k>=12 zero.
// Then posScore, fused log-softmax over 129 logits, atomic loss/acc accumulate.
// ---------------------------------------------------------------------------
__global__ __launch_bounds__(256) void score_kernel(const __bf16* __restrict__ locCb,
                                                    const __bf16* __restrict__ encB,
                                                    const float*  __restrict__ encF,
                                                    const int*    __restrict__ extIdx,
                                                    float* __restrict__ accum) {
    __shared__ float sNeg[K_ * NEG_];   // 6 KB
    __shared__ float sPos[256];         // 1 KB

    const int bw   = blockIdx.x;
    const int b    = bw / W_;
    const int w    = bw - b * W_;
    const int lane = threadIdx.x & 31;
    const int wid  = threadIdx.x >> 5;
    const int half = lane >> 4;
    const int l15  = lane & 15;

    // B matrix: column n = wid*16 + l15, gathered row
    const int n   = wid * 16 + l15;
    const int idx = extIdx[(size_t)(b * NEG_ + n) * W_ + w];
    const __bf16* bRow = encB + (size_t)idx * D_;
    __builtin_prefetch(bRow, 0, 1);

    // Gathered B row: issue all 16 b128 loads as a clause.
    v16bf bfr[8];
    #pragma unroll
    for (int s = 0; s < 8; ++s)
        bfr[s] = *(const v16bf*)(bRow + s * 32 + half * 16);

    // A matrix: row k = l15 (zero rows for k >= 12)
    const int kRow = l15;
    v16bf afr[8];
    if (kRow < K_) {
        const __bf16* aRow = locCb + (size_t)(bw * K_ + kRow) * D_;
        #pragma unroll
        for (int s = 0; s < 8; ++s) afr[s] = load_afrag(aRow, s * 32, half);
    } else {
        #pragma unroll
        for (int s = 0; s < 8; ++s) afr[s] = (v16bf){};
    }

    v8f acc = {};
    #pragma unroll
    for (int s = 0; s < 8; ++s)
        acc = __builtin_amdgcn_wmma_f32_16x16x32_bf16(
                  false, afr[s], false, bfr[s], (short)0, acc, false, false);

    // spill neg logits (scaled by 1/Denc) to LDS
    const int mBase = (lane >= 16) ? 8 : 0;
    #pragma unroll
    for (int r = 0; r < 8; ++r) {
        const int m = r + mBase;
        if (m < K_) sNeg[m * NEG_ + n] = acc[r] * (1.0f / (float)D_);
    }

    // posScore partials: thread t handles (k = t/16, e-strip g = t%16)
    const int tk = threadIdx.x >> 4;
    const int g  = threadIdx.x & 15;
    float p = 0.0f;
    if (tk < K_) {
        const __bf16* lr = locCb + (size_t)(bw * K_ + tk) * D_;
        const float*  er = encF  + (size_t)(b * S_ + (w + 1 + tk)) * D_;
        #pragma unroll
        for (int j = 0; j < 16; ++j) {
            const int e = g * 16 + j;
            p += (float)lr[e] * er[e];
        }
    }
    sPos[threadIdx.x] = p;
    __syncthreads();

    // 12 threads: finish pos, softmax over [pos | 128 negs], accumulate
    if (threadIdx.x < K_) {
        const int k = threadIdx.x;
        float pos = 0.0f;
        #pragma unroll
        for (int j = 0; j < 16; ++j) pos += sPos[k * 16 + j];
        pos *= (1.0f / (float)D_);

        const float* row = &sNeg[k * NEG_];
        float mx = pos;
        for (int q = 0; q < NEG_; ++q) mx = fmaxf(mx, row[q]);
        float se = __expf(pos - mx);
        bool win = true;
        for (int q = 0; q < NEG_; ++q) {
            const float v = row[q];
            se += __expf(v - mx);
            win = win && (pos >= v);       // argmax==0 iff pos >= all negs
        }
        const float logp0 = (pos - mx) - __logf(se);
        atomicAdd(&accum[k],      logp0);
        atomicAdd(&accum[K_ + k], win ? 1.0f : 0.0f);
    }
}

// ---------------------------------------------------------------------------
__global__ void finalize(const float* __restrict__ accum, float* __restrict__ out) {
    const int i = threadIdx.x;
    if (i < K_) {
        out[i]      = -accum[i]      * (1.0f / (float)BW_);  // losses (1,12)
        out[K_ + i] =  accum[K_ + i] * (1.0f / (float)BW_);  // acc    (1,12)
    }
}

// ---------------------------------------------------------------------------
extern "C" void kernel_launch(void* const* d_in, const int* in_sizes, int n_in,
                              void* d_out, int out_size, void* d_ws, size_t ws_size,
                              hipStream_t stream) {
    const float* cF   = (const float*)d_in[0];   // (B,S,256)
    const float* enc  = (const float*)d_in[1];   // (B,S,256)
    const float* wgt  = (const float*)d_in[2];   // (12,256,256)
    const int*   eidx = (const int*)  d_in[3];   // (B*NEG*W,)

    char* ws = (char*)d_ws;
    __bf16* encB  = (__bf16*)(ws);                             // 2 MB
    __bf16* cB    = (__bf16*)(ws + (size_t) 2 * 1024 * 1024);  // 2 MB
    __bf16* wB    = (__bf16*)(ws + (size_t) 4 * 1024 * 1024);  // 1.5 MB
    __bf16* locCb = (__bf16*)(ws + (size_t) 6 * 1024 * 1024);  // 22.8 MB
    float*  accum = (float*) (ws + (size_t)30 * 1024 * 1024);  // 24 floats

    const int nEnc = B_ * S_ * D_;       // 1,048,576
    const int nWgt = K_ * D_ * D_;       //   786,432

    cvt_f32_bf16<<<(nEnc + 255) / 256, 256, 0, stream>>>(enc, encB, nEnc);
    cvt_f32_bf16<<<(nEnc + 255) / 256, 256, 0, stream>>>(cF,  cB,   nEnc);
    cvt_f32_bf16<<<(nWgt + 255) / 256, 256, 0, stream>>>(wgt, wB,   nWgt);
    zero_accum<<<1, 32, 0, stream>>>(accum);

    locc_gemm<<<BW_ / 16, 256, 0, stream>>>(cB, wB, locCb);
    score_kernel<<<BW_, 256, 0, stream>>>(locCb, encB, enc, eidx, accum);
    finalize<<<1, 32, 0, stream>>>(accum, (float*)d_out);
}
